// Transformer_embed_ready_test_55259049230587
// MI455X (gfx1250) — matrix operand order repaired and checked by reading, not compiled, not run on hardware
//
#include <hip/hip_runtime.h>
#include <stdint.h>

// ---------------------------------------------------------------------------
// Janossy-LSTM transformer for MI455X (gfx1250, wave32, WMMA f16 16x16x32,
// async global->LDS staging with ASYNCcnt double buffering)
// ---------------------------------------------------------------------------

typedef _Float16 v16h __attribute__((ext_vector_type(16)));
typedef _Float16 v8h  __attribute__((ext_vector_type(8)));
typedef float    v8f  __attribute__((ext_vector_type(8)));

constexpr int Sq = 80;          // sequence length
constexpr int Bb = 16;          // batch
constexpr int Dm = 512;         // d_model
constexpr int Hh = 64;          // lstm hidden
constexpr int G4 = 4 * Hh;      // 256 gate columns
constexpr int Di = 2048;        // ffn inner
constexpr int Ll = 6;           // layers
constexpr int MROWS = Bb * Sq;  // 1280 activation rows

__device__ __forceinline__ float sigm(float x) { return 1.f / (1.f + __expf(-x)); }
__device__ __forceinline__ float tanh_f(float x) {
  x = fminf(fmaxf(x, -15.f), 15.f);
  float e = __expf(2.f * x);
  return (e - 1.f) / (e + 1.f);
}

// ---------------------------------------------------------------------------
// Generic WMMA GEMM:  C[M,N] = A[M,K] (f16, row-major) x W[N,K] (f16, [out,in])
// Block = 256 threads (8 waves), 64x64 tile, K-step 32.
// Staging uses GLOBAL_LOAD_ASYNC_TO_LDS_B128 with double-buffered LDS tiles:
//   issue stage k+32 -> wait asynccnt<=2 (stage k done, k+32 in flight) ->
//   barrier -> WMMA on stage k -> barrier.
// ---------------------------------------------------------------------------
template <bool BIAS, bool RELU, bool RES, bool OUT32, bool OUT16, bool REMAP>
__global__ __launch_bounds__(256) void gemm64(
    const _Float16* __restrict__ A, const _Float16* __restrict__ W,
    const float* __restrict__ bias, const float* res,
    float* out32, _Float16* out16, int M, int N, int K) {
  __shared__ _Float16 As[2][64][32];  // [buf][m][k]
  __shared__ _Float16 Ws[2][64][32];  // [buf][n][k]
  const int tid = threadIdx.x;
  const int lane = tid & 31;
  const int wave = tid >> 5;
  const int wm = wave >> 2;
  const int wn = wave & 3;
  const int bm = blockIdx.x * 64;
  const int bn = blockIdx.y * 64;

  const v8f z8 = {0.f, 0.f, 0.f, 0.f, 0.f, 0.f, 0.f, 0.f};
  v8f acc[2] = {z8, z8};

  const int la_r = tid >> 2;        // 0..63
  const int la_c = (tid & 3) * 8;   // 0,8,16,24 (halves)
  const int m16 = lane & 15;
  const int kh8 = (lane >> 4) * 8;
  const int kb16 = (lane >> 4) * 16;

  auto stage = [&](int buf, int k0) {
    // each thread moves one 16B slot of A and one of W, async into LDS
    unsigned ldsA = (unsigned)(uintptr_t)&As[buf][la_r][la_c];
    unsigned long long ga =
        (unsigned long long)(uintptr_t)&A[(size_t)(bm + la_r) * K + k0 + la_c];
    asm volatile("global_load_async_to_lds_b128 %0, %1, off"
                 :: "v"(ldsA), "v"(ga) : "memory");
    unsigned ldsW = (unsigned)(uintptr_t)&Ws[buf][la_r][la_c];
    unsigned long long gw =
        (unsigned long long)(uintptr_t)&W[(size_t)(bn + la_r) * K + k0 + la_c];
    asm volatile("global_load_async_to_lds_b128 %0, %1, off"
                 :: "v"(ldsW), "v"(gw) : "memory");
  };

  stage(0, 0);
  for (int k0 = 0; k0 < K; k0 += 32) {
    const int cur = (k0 >> 5) & 1;
    if (k0 + 32 < K) {
      stage(cur ^ 1, k0 + 32);  // safe: prev iter's trailing barrier passed
      asm volatile("s_wait_asynccnt 0x2" ::: "memory");  // cur stage complete
    } else {
      asm volatile("s_wait_asynccnt 0x0" ::: "memory");
    }
    __syncthreads();  // all waves' async data visible

    // B fragment: lane = N column (wn*16 + m16), 16 contiguous K values
    v16h bf = *(const v16h*)&Ws[cur][wn * 16 + m16][kb16];
#pragma unroll
    for (int ms = 0; ms < 2; ++ms) {
      union { v16h v; v8h h[2]; } af;
      const _Float16* ap = &As[cur][wm * 32 + ms * 16 + m16][0];
      af.v = *(const v16h*)(ap + kb16);        // K {0..15} or {16..31}
      acc[ms] = __builtin_amdgcn_wmma_f32_16x16x32_f16(
          false, af.v, false, bf, (short)0, acc[ms], false, false);
    }
    __syncthreads();  // done reading buf[cur]; next-next stage may overwrite
  }

  const int col = bn + wn * 16 + (lane & 15);
  const float bv = BIAS ? bias[col] : 0.f;
#pragma unroll
  for (int ms = 0; ms < 2; ++ms) {
#pragma unroll
    for (int r = 0; r < 8; ++r) {
      const int row = bm + wm * 32 + ms * 16 + r + 8 * (lane >> 4);
      float v = acc[ms][r] + bv;
      if (RES) v += res[(size_t)row * N + col];
      if (RELU) v = fmaxf(v, 0.f);
      const size_t orow =
          REMAP ? (size_t)((row % Sq) * Bb + row / Sq) : (size_t)row;
      if (OUT32) out32[orow * N + col] = v;
      if (OUT16) out16[orow * N + col] = (_Float16)v;
    }
  }
}

// ---------------------------------------------------------------------------
// Janossy LSTM: 80 time steps over 1280 rows (row n = i*16 + b).
// Block = 128 threads (4 waves, one per SIMD: leaves headroom for ~550 VGPRs
// per wave: 128 acc + 128 gx prefetch + compiler-cached whh B-fragments).
// One wave per output position i (16-row batch tile). h in LDS (WMMA A
// operand); c purely in registers. gx loads for step t are hoisted BEFORE the
// WMMA block so their latency hides under 32 WMMAs instead of blocking adds.
// ---------------------------------------------------------------------------
__global__ __launch_bounds__(128) void lstm_kernel(
    const float* __restrict__ gx,      // [S*B, 256], row = tp*B + b
    const _Float16* __restrict__ whh,  // [256, 64]  ([out,in] = B[N,K])
    _Float16* __restrict__ hfin) {     // [B*S, 64] f16, row = b*S + i
  __shared__ _Float16 hbuf[64][64];
  __shared__ _Float16 whs[256][64];
  const int tid = threadIdx.x;
  const int lane = tid & 31;
  const int wave = tid >> 5;            // 0..3
  const int i = blockIdx.x * 4 + wave;  // output position 0..79

  for (int t = tid; t < 64 * 64; t += 128) hbuf[t >> 6][t & 63] = (_Float16)0.f;
  for (int t = tid * 8; t < 256 * 64; t += 128 * 8)
    *(v8h*)&whs[t >> 6][t & 63] = *(const v8h*)&whh[t];
  __syncthreads();

  const v8f z8 = {0.f, 0.f, 0.f, 0.f, 0.f, 0.f, 0.f, 0.f};
  float creg[4][8];
#pragma unroll
  for (int jj = 0; jj < 4; ++jj)
#pragma unroll
    for (int r = 0; r < 8; ++r) creg[jj][r] = 0.f;

  const int m16 = lane & 15;
  const int kh8 = (lane >> 4) * 8;
  const int kb16 = (lane >> 4) * 16;
  const int arow = wave * 16 + m16;

  for (int t = 0; t < Sq; ++t) {
    const int tp = (t == i) ? (Sq - 1) : ((t == Sq - 1) ? i : t);
    const float* gxr = gx + (size_t)tp * Bb * G4;

    // ---- prefetch all 128 gate-input values for this step (independent of h)
    float garr[4][4][8];
#pragma unroll
    for (int jj = 0; jj < 4; ++jj) {
      const int col = jj * 16 + (lane & 15);
#pragma unroll
      for (int r = 0; r < 8; ++r) {
        const int b = r + 8 * (lane >> 4);
        const float* grow = gxr + (size_t)b * G4 + col;
#pragma unroll
        for (int g = 0; g < 4; ++g) garr[jj][g][r] = grow[g * 64];
      }
    }

    v8f acc[4][4];
#pragma unroll
    for (int jj = 0; jj < 4; ++jj)
#pragma unroll
      for (int g = 0; g < 4; ++g) acc[jj][g] = z8;

    // All 16 gate tiles accumulated BEFORE any h write -> single h buffer OK.
#pragma unroll
    for (int k0 = 0; k0 < 64; k0 += 32) {
      union { v16h v; v8h h[2]; } af;
      af.v = *(const v16h*)&hbuf[arow][k0 + kb16];
#pragma unroll
      for (int jj = 0; jj < 4; ++jj) {
#pragma unroll
        for (int g = 0; g < 4; ++g) {
          v16h bf = *(const v16h*)&whs[g * 64 + jj * 16 + m16][k0 + kb16];
          acc[jj][g] = __builtin_amdgcn_wmma_f32_16x16x32_f16(
              false, af.v, false, bf, (short)0, acc[jj][g], false, false);
        }
      }
    }

    // elementwise LSTM cell update (C layout: M = r + 8*(lane>>4), N = lane&15)
#pragma unroll
    for (int jj = 0; jj < 4; ++jj) {
      const int col = jj * 16 + (lane & 15);
#pragma unroll
      for (int r = 0; r < 8; ++r) {
        const int b = r + 8 * (lane >> 4);
        const float gi = acc[jj][0][r] + garr[jj][0][r];
        const float gf = acc[jj][1][r] + garr[jj][1][r];
        const float gg = acc[jj][2][r] + garr[jj][2][r];
        const float go = acc[jj][3][r] + garr[jj][3][r];
        const float cv = sigm(gf) * creg[jj][r] + sigm(gi) * tanh_f(gg);
        const float hv = sigm(go) * tanh_f(cv);
        creg[jj][r] = cv;
        hbuf[wave * 16 + b][col] = (_Float16)hv;
      }
    }
  }

  // final h -> [B, S, H] for the FC GEMM
#pragma unroll
  for (int jj = 0; jj < 4; ++jj) {
    const int col = jj * 16 + (lane & 15);
#pragma unroll
    for (int r = 0; r < 8; ++r) {
      const int b = r + 8 * (lane >> 4);
      hfin[((size_t)b * Sq + i) * Hh + col] = hbuf[wave * 16 + b][col];
    }
  }
}

// ---------------------------------------------------------------------------
// LayerNorm over D=512, one block per row, f16 output (GEMM A operand)
// ---------------------------------------------------------------------------
__global__ __launch_bounds__(256) void ln_kernel(
    const float* __restrict__ x, const float* __restrict__ g,
    const float* __restrict__ b, _Float16* __restrict__ out) {
  __shared__ float s1[8], s2[8];
  const int row = blockIdx.x, tid = threadIdx.x;
  const int lane = tid & 31, wave = tid >> 5;
  const float* xr = x + (size_t)row * Dm;
  const float v0 = xr[tid], v1 = xr[tid + 256];
  float sum = v0 + v1, sq = v0 * v0 + v1 * v1;
#pragma unroll
  for (int off = 16; off > 0; off >>= 1) {
    sum += __shfl_down(sum, off, 32);
    sq += __shfl_down(sq, off, 32);
  }
  if (lane == 0) { s1[wave] = sum; s2[wave] = sq; }
  __syncthreads();
  if (tid == 0) {
    float a = 0.f, c = 0.f;
    for (int w = 0; w < 8; ++w) { a += s1[w]; c += s2[w]; }
    s1[0] = a; s2[0] = c;
  }
  __syncthreads();
  const float mean = s1[0] * (1.f / Dm);
  const float var = s2[0] * (1.f / Dm) - mean * mean;
  const float rs = rsqrtf(var + 1e-6f);
  out[(size_t)row * Dm + tid] = (_Float16)((v0 - mean) * rs * g[tid] + b[tid]);
  out[(size_t)row * Dm + tid + 256] =
      (_Float16)((v1 - mean) * rs * g[tid + 256] + b[tid + 256]);
}

// ---------------------------------------------------------------------------
// Final LayerNorm + projection to scalar logit per row
// ---------------------------------------------------------------------------
__global__ __launch_bounds__(256) void final_kernel(
    const float* __restrict__ x, const float* __restrict__ g,
    const float* __restrict__ b, const float* __restrict__ wp,
    float* __restrict__ out) {
  __shared__ float s1[8], s2[8], s3[8];
  const int row = blockIdx.x, tid = threadIdx.x;
  const int lane = tid & 31, wave = tid >> 5;
  const float* xr = x + (size_t)row * Dm;
  const float v0 = xr[tid], v1 = xr[tid + 256];
  float sum = v0 + v1, sq = v0 * v0 + v1 * v1;
#pragma unroll
  for (int off = 16; off > 0; off >>= 1) {
    sum += __shfl_down(sum, off, 32);
    sq += __shfl_down(sq, off, 32);
  }
  if (lane == 0) { s1[wave] = sum; s2[wave] = sq; }
  __syncthreads();
  if (tid == 0) {
    float a = 0.f, c = 0.f;
    for (int w = 0; w < 8; ++w) { a += s1[w]; c += s2[w]; }
    s1[0] = a; s2[0] = c;
  }
  __syncthreads();
  const float mean = s1[0] * (1.f / Dm);
  const float var = s2[0] * (1.f / Dm) - mean * mean;
  const float rs = rsqrtf(var + 1e-6f);
  const float n0 = (v0 - mean) * rs * g[tid] + b[tid];
  const float n1 = (v1 - mean) * rs * g[tid + 256] + b[tid + 256];
  float dot = n0 * wp[tid] + n1 * wp[tid + 256];
#pragma unroll
  for (int off = 16; off > 0; off >>= 1) dot += __shfl_down(dot, off, 32);
  if (lane == 0) s3[wave] = dot;
  __syncthreads();
  if (tid == 0) {
    float a = 0.f;
    for (int w = 0; w < 8; ++w) a += s3[w];
    out[row] = a;
  }
}

// f32 -> f16 weight conversion (layout preserved: [out, in] == WMMA B [N, K])
__global__ void cvt_f16(const float* __restrict__ in, _Float16* __restrict__ out,
                        int n) {
  const int idx = blockIdx.x * 256 + threadIdx.x;
  if (idx < n) out[idx] = (_Float16)in[idx];
}

// ---------------------------------------------------------------------------
extern "C" void kernel_launch(void* const* d_in, const int* in_sizes, int n_in,
                              void* d_out, int out_size, void* d_ws,
                              size_t ws_size, hipStream_t stream) {
  const float* src   = (const float*)d_in[0];
  // d_in[1] = trg_seq (unused by the reference)
  const float* ln1_g = (const float*)d_in[2];
  const float* ln1_b = (const float*)d_in[3];
  const float* wih   = (const float*)d_in[4];   // [L, 256, 512]
  const float* whh   = (const float*)d_in[5];   // [L, 256, 64]
  const float* wfc   = (const float*)d_in[6];   // [L, 512, 64]
  const float* ln2_g = (const float*)d_in[7];
  const float* ln2_b = (const float*)d_in[8];
  const float* w1    = (const float*)d_in[9];   // [L, 2048, 512]
  const float* b1    = (const float*)d_in[10];
  const float* w2    = (const float*)d_in[11];  // [L, 512, 2048]
  const float* b2    = (const float*)d_in[12];
  const float* lnf_g = (const float*)d_in[13];
  const float* lnf_b = (const float*)d_in[14];
  const float* wprj  = (const float*)d_in[15];  // [1, 512]

  char* base = (char*)d_ws;
  size_t off = 0;
  auto wsa = [&](size_t bytes) -> void* {
    void* p = base + off;
    off += (bytes + 255) & ~(size_t)255;
    return p;
  };
  _Float16* wih_h = (_Float16*)wsa((size_t)Ll * G4 * Dm * 2);
  _Float16* whh_h = (_Float16*)wsa((size_t)Ll * G4 * Hh * 2);
  _Float16* wfc_h = (_Float16*)wsa((size_t)Ll * Dm * Hh * 2);
  _Float16* w1_h  = (_Float16*)wsa((size_t)Ll * Di * Dm * 2);
  _Float16* w2_h  = (_Float16*)wsa((size_t)Ll * Dm * Di * 2);
  float*    xbuf  = (float*)wsa((size_t)MROWS * Dm * 4);
  _Float16* xln   = (_Float16*)wsa((size_t)MROWS * Dm * 2);
  float*    xg    = (float*)wsa((size_t)MROWS * G4 * 4);
  _Float16* hfin  = (_Float16*)wsa((size_t)MROWS * Hh * 2);
  _Float16* y1    = (_Float16*)wsa((size_t)MROWS * Di * 2);

  auto cvt = [&](const float* in, _Float16* o, int n) {
    cvt_f16<<<(n + 255) / 256, 256, 0, stream>>>(in, o, n);
  };
  cvt(wih, wih_h, Ll * G4 * Dm);
  cvt(whh, whh_h, Ll * G4 * Hh);
  cvt(wfc, wfc_h, Ll * Dm * Hh);
  cvt(w1, w1_h, Ll * Di * Dm);
  cvt(w2, w2_h, Ll * Dm * Di);

  hipMemcpyAsync(xbuf, src, (size_t)MROWS * Dm * sizeof(float),
                 hipMemcpyDeviceToDevice, stream);

  for (int l = 0; l < Ll; ++l) {
    // x -> LN -> f16
    ln_kernel<<<MROWS, 256, 0, stream>>>(xbuf, ln1_g + l * Dm, ln1_b + l * Dm,
                                         xln);
    // gate projections: xg[s*B+b, 256] = LN(x)[b*S+s] @ wih^T   (REMAP rows)
    gemm64<false, false, false, true, false, true>
        <<<dim3(MROWS / 64, G4 / 64), 256, 0, stream>>>(
            xln, wih_h + (size_t)l * G4 * Dm, nullptr, nullptr, xg, nullptr,
            MROWS, G4, Dm);
    // Janossy LSTM (the sequential bottleneck)
    lstm_kernel<<<Sq / 4, 128, 0, stream>>>(
        xg, whh_h + (size_t)l * G4 * Hh, hfin);
    // x = h @ wfc^T + x
    gemm64<false, false, true, true, false, false>
        <<<dim3(MROWS / 64, Dm / 64), 256, 0, stream>>>(
            hfin, wfc_h + (size_t)l * Dm * Hh, nullptr, xbuf, xbuf, nullptr,
            MROWS, Dm, Hh);
    // FFN
    ln_kernel<<<MROWS, 256, 0, stream>>>(xbuf, ln2_g + l * Dm, ln2_b + l * Dm,
                                         xln);
    gemm64<true, true, false, false, true, false>
        <<<dim3(MROWS / 64, Di / 64), 256, 0, stream>>>(
            xln, w1_h + (size_t)l * Di * Dm, b1 + l * Di, nullptr, nullptr, y1,
            MROWS, Di, Dm);
    gemm64<true, false, true, true, false, false>
        <<<dim3(MROWS / 64, Dm / 64), 256, 0, stream>>>(
            y1, w2_h + (size_t)l * Dm * Di, b2 + l * Dm, xbuf, xbuf, nullptr,
            MROWS, Dm, Di);
  }

  final_kernel<<<MROWS, 256, 0, stream>>>(xbuf, lnf_g, lnf_b, wprj,
                                          (float*)d_out);
}
